// DenseAttentionAggregator_46033459479181
// MI455X (gfx1250) — matrix operand order repaired and checked by reading, not compiled, not run on hardware
//
#include <hip/hip_runtime.h>

typedef __attribute__((ext_vector_type(16))) _Float16 v16h;
typedef __attribute__((ext_vector_type(8)))  _Float16 v8h;
typedef __attribute__((ext_vector_type(8)))  float    v8f;

#define D_IN    512
#define D_HID   32
#define D_OUT   256
#define N_X     8192
#define N_NEIGH 8192
#define NEG_BIG_F (-9.0e15f)

// ---------------------------------------------------------------------------
// att = tanh(X @ W1 + b1) @ W2 + b2   ->  f16 (M x 32)
// 256 threads = 8 rows x 32 hidden cols per block.
// ---------------------------------------------------------------------------
__global__ void __launch_bounds__(256)
mlp_att_kernel(const float* __restrict__ X, const float* __restrict__ W1,
               const float* __restrict__ b1, const float* __restrict__ W2,
               const float* __restrict__ b2, _Float16* __restrict__ att) {
    __shared__ float h[8][D_HID];
    const int t = threadIdx.x;
    const int r = t >> 5;          // 0..7
    const int c = t & 31;          // 0..31
    const int row = blockIdx.x * 8 + r;
    const float* xp = X + (size_t)row * D_IN;
    float acc = b1[c];
    #pragma unroll 4
    for (int k = 0; k < D_IN; ++k) acc += xp[k] * W1[k * D_HID + c];
    h[r][c] = tanhf(acc);
    __syncthreads();
    float acc2 = b2[c];
    #pragma unroll
    for (int k = 0; k < D_HID; ++k) acc2 += h[r][k] * W2[k * D_HID + c];
    att[(size_t)row * D_HID + c] = (_Float16)acc2;
}

// ---------------------------------------------------------------------------
// WT[n*K + k] = (f16) W[k*N + n]   (weight transpose + f32->f16 convert)
// ---------------------------------------------------------------------------
__global__ void __launch_bounds__(256)
transpose_cvt_kernel(const float* __restrict__ W, _Float16* __restrict__ WT,
                     int K, int N) {
    int i = blockIdx.x * blockDim.x + threadIdx.x;
    if (i < K * N) {
        int k = i / N, n = i % N;
        WT[(size_t)n * K + k] = (_Float16)W[i];
    }
}

// ---------------------------------------------------------------------------
// C = A(MxK f32) x B(KxN) + bias, B given transposed as f16 (N x K).
// One wave = one 16x16 tile via v_wmma_f32_16x16x32_f16, K-step 32.
// outT   (optional): f16, transposed (N x M)       -- used for valueT
// outF32 (optional): f32, row stride out_stride    -- used for x@Wfx half
// ---------------------------------------------------------------------------
__global__ void __launch_bounds__(256)
gemm_wmma_kernel(const float* __restrict__ A, const _Float16* __restrict__ BT,
                 const float* __restrict__ bias,
                 _Float16* __restrict__ outT, float* __restrict__ outF32,
                 int M, int N, int K, int out_stride) {
    const int wid  = threadIdx.x >> 5;
    const int lane = threadIdx.x & 31;
    const int half = lane >> 4;
    const int lr   = lane & 15;

    const int tiles_n = N >> 4;
    const int tile = blockIdx.x * 8 + wid;
    const int r0 = (tile / tiles_n) << 4;
    const int n0 = (tile % tiles_n) << 4;

    v8f c = {};
    for (int k = 0; k < K; k += 32) {
        // A fragment: row r0+lr, contiguous K range 16h..16h+15
        const float* ap = A + (size_t)(r0 + lr) * K + k + 16 * half;
        v16h a;
        #pragma unroll
        for (int i = 0; i < 8; ++i) {
            a[i]     = (_Float16)ap[i];
            a[8 + i] = (_Float16)ap[8 + i];
        }
        // B fragment: col n0+lr, contiguous K range 16h..16h+15
        v16h b = *(const v16h*)(BT + (size_t)(n0 + lr) * K + k + 16 * half);
        c = __builtin_amdgcn_wmma_f32_16x16x32_f16(false, a, false, b,
                                                   (short)0, c, false, false);
    }
    const float bb = bias[n0 + lr];
    #pragma unroll
    for (int v = 0; v < 8; ++v) {
        const int Mr = v + 8 * half;       // C layout: M = v + 8*half, N = lr
        const float val = c[v] + bb;
        if (outF32) outF32[(size_t)(r0 + Mr) * out_stride + n0 + lr] = val;
        if (outT)   outT[(size_t)(n0 + lr) * M + r0 + Mr] = (_Float16)val;
    }
}

// ---------------------------------------------------------------------------
// Async staging of one 16x32 int32 adj tile (2 KB) into LDS.
// 128 threads x one b128 each; tracked by ASYNCcnt.
// lds_dst = byte offset of destination within the workgroup LDS allocation
// (low 32 bits of a flat pointer to __shared__).
// ---------------------------------------------------------------------------
__device__ __forceinline__ void async_load_adj_tile(const int* __restrict__ adj,
                                                    int r0, int j0,
                                                    unsigned lds_dst_base, int t) {
    const int row = t >> 3;             // 0..15
    const int cg  = (t & 7) << 2;       // 0,4,...,28
    const int* g = adj + (size_t)(r0 + row) * N_NEIGH + j0 + cg;
    const unsigned dst = lds_dst_base + (unsigned)(t << 4);
    asm volatile("global_load_async_to_lds_b128 %0, %1, off"
                 :: "v"(dst), "v"(g) : "memory");
}

// ---------------------------------------------------------------------------
// Flash-attention aggregation. Block = 16 x-rows, 4 waves; wave w owns
// output columns [w*64, w*64+64). Single pass over neigh: adj read exactly
// once, streamed via async-to-LDS two tiles ahead (triple buffered).
// ---------------------------------------------------------------------------
__global__ void __launch_bounds__(128)
attn_flash_kernel(const _Float16* __restrict__ xatt,
                  const _Float16* __restrict__ natt,
                  const int* __restrict__ adj,
                  const _Float16* __restrict__ valueT,
                  float* __restrict__ out) {
    __shared__ __align__(32) _Float16 Pbuf[4][16][32];   // 4 KB
    __shared__ __align__(16) int adjBuf[3][16][32];      // 6 KB, triple buffer

    const int t    = threadIdx.x;
    const int wid  = t >> 5;
    const int lane = t & 31;
    const int half = lane >> 4;
    const int lr   = lane & 15;
    const int r0   = blockIdx.x << 4;
    const int ncol = wid * 64;

    const unsigned adj_lds_base = (unsigned)(size_t)(void*)&adjBuf[0][0][0];

    // Resident A fragment: x_att rows r0..r0+15 (16x32 f16)
    v16h axa;
    {
        const _Float16* p = xatt + (size_t)(r0 + lr) * D_HID + 16 * half;
        v8h lo = *(const v8h*)p;
        v8h hi = *(const v8h*)(p + 8);
        #pragma unroll
        for (int i = 0; i < 8; ++i) { axa[i] = lo[i]; axa[8 + i] = hi[i]; }
    }

    float mrow[8], lrow[8];
    #pragma unroll
    for (int v = 0; v < 8; ++v) { mrow[v] = -3.0e38f; lrow[v] = 0.0f; }
    v8f o0 = {}, o1 = {}, o2 = {}, o3 = {};

    // Prologue: prefetch tiles 0 and 1
    async_load_adj_tile(adj, r0, 0,  adj_lds_base,        t);
    async_load_adj_tile(adj, r0, 32, adj_lds_base + 2048, t);

    #pragma unroll 1
    for (int j0 = 0; j0 < N_NEIGH; j0 += 32) {
        const int it  = j0 >> 5;
        const int buf = it % 3;

        // Tile `it` ready when own-wave ASYNCcnt drops to (#tiles issued after it)
        if (j0 + 32 < N_NEIGH) asm volatile("s_wait_asynccnt 1" ::: "memory");
        else                   asm volatile("s_wait_asynccnt 0" ::: "memory");
        __syncthreads();   // all waves' async data visible; prev Pbuf reads done

        // Prefetch tile it+2 into the buffer freed at the barrier above
        if (j0 + 64 < N_NEIGH)
            async_load_adj_tile(adj, r0, j0 + 64,
                                adj_lds_base + (unsigned)((it + 2) % 3) * 2048, t);

        // ---- S = x_att @ neib_att^T for 32 neighbor columns (2 tiles) ----
        v16h nb0 = *(const v16h*)(natt + (size_t)(j0 + lr) * D_HID + 16 * half);
        v16h nb1 = *(const v16h*)(natt + (size_t)(j0 + 16 + lr) * D_HID + 16 * half);
        v8f cz = {};
        v8f s0 = __builtin_amdgcn_wmma_f32_16x16x32_f16(false, axa, false, nb0,
                                                        (short)0, cz, false, false);
        v8f s1 = __builtin_amdgcn_wmma_f32_16x16x32_f16(false, axa, false, nb1,
                                                        (short)0, cz, false, false);

        // ---- leaky-relu, adj mask (from LDS), per-row tile max ----
        const int* abuf = (const int*)adjBuf[buf];
        float p0[8], p1[8], tmax[8];
        #pragma unroll
        for (int v = 0; v < 8; ++v) {
            const int Mr = v + 8 * half;
            float a0 = s0[v]; a0 = a0 > 0.f ? a0 : 0.01f * a0;
            float a1 = s1[v]; a1 = a1 > 0.f ? a1 : 0.01f * a1;
            a0 = (abuf[Mr * 32 + lr]      > 0) ? a0 : NEG_BIG_F;
            a1 = (abuf[Mr * 32 + 16 + lr] > 0) ? a1 : NEG_BIG_F;
            p0[v] = a0; p1[v] = a1;
            float tm = fmaxf(a0, a1);
            #pragma unroll
            for (int off = 1; off < 16; off <<= 1)
                tm = fmaxf(tm, __shfl_xor(tm, off, 32));
            tmax[v] = tm;
        }

        // ---- online softmax update; stage P (f16, A-layout source) in LDS ----
        #pragma unroll
        for (int v = 0; v < 8; ++v) {
            const float mnew  = fmaxf(mrow[v], tmax[v]);
            const float scale = __expf(mrow[v] - mnew);
            const float e0 = __expf(p0[v] - mnew);
            const float e1 = __expf(p1[v] - mnew);
            float rsum = e0 + e1;
            #pragma unroll
            for (int off = 1; off < 16; off <<= 1)
                rsum += __shfl_xor(rsum, off, 32);
            lrow[v] = lrow[v] * scale + rsum;
            mrow[v] = mnew;
            o0[v] *= scale; o1[v] *= scale; o2[v] *= scale; o3[v] *= scale;
            const int Mr = v + 8 * half;
            Pbuf[wid][Mr][lr]      = (_Float16)e0;
            Pbuf[wid][Mr][16 + lr] = (_Float16)e1;
        }
        __syncthreads();   // Pbuf ready

        // ---- reload P as 16x32 A fragment ----
        v16h pa;
        {
            const _Float16* p = &Pbuf[wid][lr][16 * half];
            v8h lo = *(const v8h*)p;
            v8h hi = *(const v8h*)(p + 8);
            #pragma unroll
            for (int i = 0; i < 8; ++i) { pa[i] = lo[i]; pa[8 + i] = hi[i]; }
        }

        // ---- O += P @ value for this wave's 64 output columns ----
        const _Float16* vb = valueT + (size_t)(ncol + lr) * N_NEIGH + j0 + 16 * half;
        v16h b0 = *(const v16h*)(vb);
        v16h b1 = *(const v16h*)(vb + (size_t)16 * N_NEIGH);
        v16h b2 = *(const v16h*)(vb + (size_t)32 * N_NEIGH);
        v16h b3 = *(const v16h*)(vb + (size_t)48 * N_NEIGH);
        o0 = __builtin_amdgcn_wmma_f32_16x16x32_f16(false, pa, false, b0, (short)0, o0, false, false);
        o1 = __builtin_amdgcn_wmma_f32_16x16x32_f16(false, pa, false, b1, (short)0, o1, false, false);
        o2 = __builtin_amdgcn_wmma_f32_16x16x32_f16(false, pa, false, b2, (short)0, o2, false, false);
        o3 = __builtin_amdgcn_wmma_f32_16x16x32_f16(false, pa, false, b3, (short)0, o3, false, false);
    }

    // ---- normalize and store right half of output (cols 256..511) ----
    #pragma unroll
    for (int v = 0; v < 8; ++v) {
        const int Mr = v + 8 * half;
        const float inv = 1.0f / lrow[v];
        float* op = out + (size_t)(r0 + Mr) * (2 * D_OUT) + D_OUT + ncol + lr;
        op[0]  = o0[v] * inv;
        op[16] = o1[v] * inv;
        op[32] = o2[v] * inv;
        op[48] = o3[v] * inv;
    }
}

// ---------------------------------------------------------------------------
extern "C" void kernel_launch(void* const* d_in, const int* in_sizes, int n_in,
                              void* d_out, int out_size, void* d_ws, size_t ws_size,
                              hipStream_t stream) {
    const float* x     = (const float*)d_in[0];
    const float* neigh = (const float*)d_in[1];
    const int*   adj   = (const int*)  d_in[2];
    const float* Wx1   = (const float*)d_in[3];
    const float* bx1   = (const float*)d_in[4];
    const float* Wx2   = (const float*)d_in[5];
    const float* bx2   = (const float*)d_in[6];
    const float* Wn1   = (const float*)d_in[7];
    const float* bn1   = (const float*)d_in[8];
    const float* Wn2   = (const float*)d_in[9];
    const float* bn2   = (const float*)d_in[10];
    const float* Wv    = (const float*)d_in[11];
    const float* bv    = (const float*)d_in[12];
    const float* Wfx   = (const float*)d_in[13];
    const float* bfx   = (const float*)d_in[14];
    float* out = (float*)d_out;

    char* ws = (char*)d_ws;
    _Float16* xatt   = (_Float16*)(ws);                  // 8192*32 f16 = 512 KB
    _Float16* natt   = (_Float16*)(ws + 524288);         // 512 KB
    _Float16* valueT = (_Float16*)(ws + 1048576);        // 256*8192 f16 = 4 MB
    _Float16* WvT    = (_Float16*)(ws + 5242880);        // 256*512 f16 = 256 KB
    _Float16* WfxT   = (_Float16*)(ws + 5505024);        // 256 KB

    // attention feature MLPs
    mlp_att_kernel<<<N_X / 8,     256, 0, stream>>>(x,     Wx1, bx1, Wx2, bx2, xatt);
    mlp_att_kernel<<<N_NEIGH / 8, 256, 0, stream>>>(neigh, Wn1, bn1, Wn2, bn2, natt);

    // weight transposes (f32 -> f16, N x K)
    const int wElems = D_IN * D_OUT;
    transpose_cvt_kernel<<<(wElems + 255) / 256, 256, 0, stream>>>(Wv,  WvT,  D_IN, D_OUT);
    transpose_cvt_kernel<<<(wElems + 255) / 256, 256, 0, stream>>>(Wfx, WfxT, D_IN, D_OUT);

    // value = neigh @ Wv + bv  (stored transposed f16 for PV B-fragments)
    gemm_wmma_kernel<<<(N_NEIGH / 16) * (D_OUT / 16) / 8, 256, 0, stream>>>(
        neigh, WvT, bv, valueT, nullptr, N_NEIGH, D_OUT, D_IN, 0);

    // left half of output: x @ Wfx + bfx  (f32, row stride 512)
    gemm_wmma_kernel<<<(N_X / 16) * (D_OUT / 16) / 8, 256, 0, stream>>>(
        x, WfxT, bfx, nullptr, out, N_X, D_OUT, D_IN, 2 * D_OUT);

    // fused masked-softmax attention aggregation -> right half of output
    attn_flash_kernel<<<N_X / 16, 128, 0, stream>>>(xatt, natt, adj, valueT, out);
}
